// ARMNetModel_47347719471425
// MI455X (gfx1250) — matrix operand order repaired and checked by reading, not compiled, not run on hardware
//
#include <hip/hip_runtime.h>
#include <cstddef>
#include <cstdint>

// ---------------- problem constants ----------------
constexpr int BSZ      = 2048;
constexpr int NFIELD   = 100;
constexpr int NEMB     = 64;
constexpr int DK       = 64;
constexpr int NHID     = 256;
constexpr int MLP_NHID = 1024;
constexpr int FPAD     = 128;           // NFIELD padded to WMMA K multiple
constexpr float EPS    = 1e-5f;

// ---------------- WMMA types ----------------
typedef __bf16 bf16_t;
typedef __attribute__((ext_vector_type(16))) __bf16 v16bf;
typedef __attribute__((ext_vector_type(8)))  __bf16 v8bf;
typedef __attribute__((ext_vector_type(8)))  float  v8f;
typedef int v4i_vs __attribute__((vector_size(16)));   // matches async-builtin param

#if __has_builtin(__builtin_amdgcn_global_load_async_to_lds_b128)
#define ASYNC_COPY 1
#endif

__device__ inline void wait_async_lds() {
#ifdef ASYNC_COPY
#if __has_builtin(__builtin_amdgcn_s_wait_asynccnt)
  __builtin_amdgcn_s_wait_asynccnt(0);
#else
  asm volatile("s_wait_asynccnt 0" ::: "memory");
#endif
#endif
}

__device__ inline v8f v8f_zero() {
  v8f z;
#pragma unroll
  for (int i = 0; i < 8; ++i) z[i] = 0.0f;
  return z;
}

__device__ inline bf16_t f2bf(float f) {
  union { float f; unsigned u; } a; a.f = f;
  unsigned r = a.u + 0x7FFFu + ((a.u >> 16) & 1u);     // RNE
  union { unsigned short s; bf16_t b; } c; c.s = (unsigned short)(r >> 16);
  return c.b;
}
__device__ inline float bf2f(bf16_t b) {
  union { unsigned short s; bf16_t b; } c; c.b = b;
  union { unsigned u; float f; } a; a.u = ((unsigned)c.s) << 16;
  return a.f;
}
__device__ inline bf16_t bf_zero() {
  union { unsigned short s; bf16_t b; } c; c.s = 0; return c.b;
}

// A fragment, 16x32 bf16: lane row m holds K=kb..kb+15 contiguous
// with kb = (lane>=16)?16:0.  Two 16B contiguous loads.
__device__ inline v16bf load_a_frag(const bf16_t* Arows, int lda, int k0, int lane) {
  int m  = lane & 15;
  int kb = (lane >> 4) << 4;                 // 0 or 16
  const bf16_t* p = Arows + (size_t)m * lda + k0 + kb;
  v8bf lo = *(const v8bf*)p;
  v8bf hi = *(const v8bf*)(p + 8);
  v16bf r;
#pragma unroll
  for (int i = 0; i < 8; ++i) { r[i] = lo[i]; r[8 + i] = hi[i]; }
  return r;
}

// B fragment from B^T (row-major N x K): lane holds column n = lane&15,
// K = (lane<16 ? 0..15 : 16..31) contiguous -> one 32B load.
__device__ inline v16bf load_b_frag(const bf16_t* Btrows, int ldb, int k0, int lane) {
  int n  = lane & 15;
  int kb = (lane >> 4) << 4;                 // 0 or 16
  return *(const v16bf*)(Btrows + (size_t)n * ldb + k0 + kb);
}

// guarded variant (rows >= nmax read as zero)
__device__ inline v16bf load_b_frag_guard(const bf16_t* Bt, int ldb, int n0, int k0,
                                          int nmax, int lane) {
  int n  = n0 + (lane & 15);
  int kb = (lane >> 4) << 4;
  if (n < nmax) return *(const v16bf*)(Bt + (size_t)n * ldb + k0 + kb);
  v16bf z;
#pragma unroll
  for (int i = 0; i < 16; ++i) z[i] = bf_zero();
  return z;
}

__device__ inline v8f wmma_bf16(v16bf a, v16bf b, v8f c) {
  return __builtin_amdgcn_wmma_f32_16x16x32_bf16(false, a, false, b, (short)0, c,
                                                 false, false);
}

// wave32 reductions
__device__ inline float wave_sum(float v) {
#pragma unroll
  for (int m = 16; m >= 1; m >>= 1) v += __shfl_xor(v, m, 32);
  return v;
}
__device__ inline float wave_max(float v) {
#pragma unroll
  for (int m = 16; m >= 1; m >>= 1) v = fmaxf(v, __shfl_xor(v, m, 32));
  return v;
}

// ---------------- kernel 0a: Wc = (query @ w_bilinear) * DK^-0.5, bf16 ----------------
__global__ void k_wc(const float* __restrict__ q, const float* __restrict__ wb,
                     bf16_t* __restrict__ wc) {
  int i = blockIdx.x * blockDim.x + threadIdx.x;
  if (i >= NHID * NEMB) return;
  int o = i >> 6, e = i & 63;
  float s = 0.f;
#pragma unroll 8
  for (int k = 0; k < DK; ++k) s += q[o * DK + k] * wb[k * NEMB + e];
  wc[i] = f2bf(s * 0.125f);                  // 1/sqrt(64)
}

// ---------------- kernel 0b: gather + clip, write emb [b,f,64] and embT [b,64,128] ----
__global__ void k_gather(const int* __restrict__ ids, const float* __restrict__ values,
                         const float* __restrict__ table, bf16_t* __restrict__ emb,
                         bf16_t* __restrict__ embT) {
  long i = (long)blockIdx.x * blockDim.x + threadIdx.x;
  if (i >= (long)BSZ * NFIELD * NEMB) return;
  int e  = (int)(i & 63);
  long bf = i >> 6;
  int f  = (int)(bf % NFIELD);
  int b  = (int)(bf / NFIELD);
  float v = values[bf];
  v = fminf(fmaxf(v, 0.001f), 1.0f);
  float x = table[(size_t)ids[bf] * NEMB + e] * v;
  bf16_t h = f2bf(x);
  emb[i] = h;
  embT[((size_t)(b * NEMB + e) << 7) + f] = h;
}

// ---------------- kernel: fused gates GEMM + entmax + att_values -> arm_w -------------
// block: 256 thr = 8 waves; tile M(o)=128 x N(f)=112(>=100), K=64
__global__ __launch_bounds__(256) void k_gates_entmax(
    const bf16_t* __restrict__ Wc, const bf16_t* __restrict__ emb,
    const float* __restrict__ attv, bf16_t* __restrict__ armw) {
  __shared__ float gt[128][112];
  int b    = blockIdx.x >> 1;
  int o0   = (blockIdx.x & 1) << 7;
  int lane = threadIdx.x & 31;
  int w    = threadIdx.x >> 5;
  const bf16_t* embB = emb + (size_t)b * NFIELD * NEMB;
  const bf16_t* Arow = Wc + (size_t)(o0 + (w << 4)) * NEMB;

  v8f acc[7];
#pragma unroll
  for (int t = 0; t < 7; ++t) acc[t] = v8f_zero();

#pragma unroll
  for (int k0 = 0; k0 < NEMB; k0 += 32) {
    v16bf a = load_a_frag(Arow, NEMB, k0, lane);
#pragma unroll
    for (int nt = 0; nt < 7; ++nt) {
      v16bf bfr = load_b_frag_guard(embB, NEMB, nt << 4, k0, NFIELD, lane);
      acc[nt] = wmma_bf16(a, bfr, acc[nt]);
    }
  }
  int hi = lane >> 4, n = lane & 15;
#pragma unroll
  for (int nt = 0; nt < 7; ++nt)
#pragma unroll
    for (int r = 0; r < 8; ++r)
      gt[(w << 4) + (hi << 3) + r][(nt << 4) + n] = acc[nt][r];
  __syncthreads();

  // entmax (alpha=1.5) over f, one wave per row, 30 bisection iterations
  for (int row = w; row < 128; row += 8) {
    float xv[4];
#pragma unroll
    for (int j = 0; j < 4; ++j) {
      int f = lane + (j << 5);
      xv[j] = (f < NFIELD) ? gt[row][f] * 0.5f : -3.0e38f;   // x*(alpha-1)
    }
    float mx = fmaxf(fmaxf(xv[0], xv[1]), fmaxf(xv[2], xv[3]));
    mx = wave_max(mx);
    float tau = mx - 1.0f;
    float dm  = 1.0f - 0.1f;                 // tau_hi - tau_lo, (1/100)^0.5 = 0.1
    for (int it = 0; it < 30; ++it) {
      dm *= 0.5f;
      float tm = tau + dm, s = 0.f;
#pragma unroll
      for (int j = 0; j < 4; ++j) {
        float d = fmaxf(xv[j] - tm, 0.f);
        s += d * d;
      }
      s = wave_sum(s);
      if (s >= 1.0f) tau = tm;
    }
    float p[4], ps = 0.f;
#pragma unroll
    for (int j = 0; j < 4; ++j) {
      float d = fmaxf(xv[j] - tau, 0.f);
      p[j] = d * d; ps += p[j];
    }
    ps = wave_sum(ps);
    float inv = 1.0f / ps;
    int o = o0 + row;
    bf16_t* dst = armw + ((size_t)(b * NHID + o) << 7);
#pragma unroll
    for (int j = 0; j < 4; ++j) {
      int f = lane + (j << 5);
      float pv = (f < NFIELD) ? p[j] * inv * attv[o * NFIELD + f] : 0.f;
      dst[f] = f2bf(pv);                     // also zero-fills pad f in [100,128)
    }
  }
}

// ---------------- kernel: arm = exp(armw @ embT^T) per batch, M=256 N=64 K=128 --------
__global__ __launch_bounds__(256) void k_arm(const bf16_t* __restrict__ armw,
                                             const bf16_t* __restrict__ embT,
                                             float* __restrict__ arm) {
  int b = blockIdx.x;
  int lane = threadIdx.x & 31, w = threadIdx.x >> 5;
  const bf16_t* A  = armw + ((size_t)b << 15);   // 256*128
  const bf16_t* Bt = embT + ((size_t)b << 13);   // 64*128
  v8f acc[2][4];
#pragma unroll
  for (int mt = 0; mt < 2; ++mt)
#pragma unroll
    for (int nt = 0; nt < 4; ++nt) acc[mt][nt] = v8f_zero();

#pragma unroll
  for (int k0 = 0; k0 < FPAD; k0 += 32) {
    v16bf a0 = load_a_frag(A + (size_t)((w << 5) + 0)  * FPAD, FPAD, k0, lane);
    v16bf a1 = load_a_frag(A + (size_t)((w << 5) + 16) * FPAD, FPAD, k0, lane);
#pragma unroll
    for (int nt = 0; nt < 4; ++nt) {
      v16bf bfr = load_b_frag(Bt + (size_t)(nt << 4) * FPAD, FPAD, k0, lane);
      acc[0][nt] = wmma_bf16(a0, bfr, acc[0][nt]);
      acc[1][nt] = wmma_bf16(a1, bfr, acc[1][nt]);
    }
  }
  int hi = lane >> 4, n = lane & 15;
  float* C = arm + ((size_t)b << 14);            // 256*64
#pragma unroll
  for (int mt = 0; mt < 2; ++mt)
#pragma unroll
    for (int nt = 0; nt < 4; ++nt)
#pragma unroll
      for (int r = 0; r < 8; ++r) {
        int o = (w << 5) + (mt << 4) + (hi << 3) + r;
        C[(size_t)o * NEMB + (nt << 4) + n] = expf(acc[mt][nt][r]);
      }
}

// ---------------- generic bf16 WMMA GEMM: C = A @ Bt^T + bias, block 128x128 ----------
// B tile staged in LDS via async copy (double-buffered); A direct from global.
constexpr int BSTRIDE = 40;                      // padded LDS row stride (bf16), 80B
__global__ __launch_bounds__(256) void k_gemm(const bf16_t* __restrict__ A,
                                              const bf16_t* __restrict__ Bt,
                                              const float* __restrict__ bias,
                                              float* __restrict__ C,
                                              int M, int N, int K) {
  __shared__ __align__(32) bf16_t sB[2][128 * BSTRIDE];   // 2 x 10 KB
  int lane = threadIdx.x & 31, w = threadIdx.x >> 5;
  int m0 = (blockIdx.x << 7) + (w << 4);
  int n0 = blockIdx.y << 7;
  const bf16_t* Arow = A + (size_t)m0 * K;

  // staging assignment: thread t copies 32B of row t>>1, half t&1
  int srow = threadIdx.x >> 1, shalf = (threadIdx.x & 1) << 4;
  const bf16_t* gRow = Bt + (size_t)(n0 + srow) * K + shalf;

  v8f acc[8];
#pragma unroll
  for (int nt = 0; nt < 8; ++nt) acc[nt] = v8f_zero();

  auto stage = [&](int k0, int buf) {
    const bf16_t* src = gRow + k0;
    bf16_t* dst = &sB[buf][srow * BSTRIDE + shalf];
#ifdef ASYNC_COPY
    __builtin_amdgcn_global_load_async_to_lds_b128((v4i_vs*)src, (v4i_vs*)dst, 0, 0);
    __builtin_amdgcn_global_load_async_to_lds_b128((v4i_vs*)(src + 8),
                                                   (v4i_vs*)(dst + 8), 0, 0);
#else
    v8bf t0 = *(const v8bf*)src;
    v8bf t1 = *(const v8bf*)(src + 8);
    *(v8bf*)dst = t0;
    *(v8bf*)(dst + 8) = t1;
#endif
  };

  stage(0, 0);
  int buf = 0;
  for (int k0 = 0; k0 < K; k0 += 32) {
    wait_async_lds();
    __syncthreads();                       // staged tile visible; prev readers done
    if (k0 + 32 < K) stage(k0 + 32, buf ^ 1);
    __builtin_prefetch((const void*)(Arow + (size_t)(lane & 15) * K + k0 + 256), 0, 0);
    v16bf a = load_a_frag(Arow, K, k0, lane);
    const bf16_t* sbase = sB[buf];
    int nn = lane & 15, kb = (lane >> 4) << 4;
#pragma unroll
    for (int nt = 0; nt < 8; ++nt) {
      const bf16_t* p = sbase + (size_t)((nt << 4) + nn) * BSTRIDE + kb;
      v8bf lo = *(const v8bf*)p;
      v8bf hi = *(const v8bf*)(p + 8);
      v16bf b;
#pragma unroll
      for (int i = 0; i < 8; ++i) { b[i] = lo[i]; b[8 + i] = hi[i]; }
      acc[nt] = wmma_bf16(a, b, acc[nt]);
    }
    buf ^= 1;
  }

  int hi = lane >> 4, n = lane & 15;
#pragma unroll
  for (int nt = 0; nt < 8; ++nt) {
    int col = n0 + (nt << 4) + n;
    float bv = bias ? bias[col] : 0.f;
#pragma unroll
    for (int r = 0; r < 8; ++r) {
      int row = m0 + (hi << 3) + r;
      C[(size_t)row * N + col] = acc[nt][r] + bv;
    }
  }
}

// ---------------- BN stats for arm: per-o mean/var over (b,e) ----------------
__global__ __launch_bounds__(256) void k_stats_arm(const float* __restrict__ arm,
                                                   float* __restrict__ mean,
                                                   float* __restrict__ var) {
  __shared__ float ss[256], sq[256];
  int o = blockIdx.x, t = threadIdx.x;
  float s = 0.f, q = 0.f;
  for (int i = t; i < BSZ * NEMB; i += 256) {
    int b = i >> 6, e = i & 63;
    float v = arm[(((size_t)b * NHID + o) << 6) + e];
    s += v; q += v * v;
  }
  ss[t] = s; sq[t] = q; __syncthreads();
  for (int off = 128; off; off >>= 1) {
    if (t < off) { ss[t] += ss[t + off]; sq[t] += sq[t + off]; }
    __syncthreads();
  }
  if (t == 0) {
    float cnt = (float)(BSZ * NEMB);
    float m = ss[0] / cnt;
    mean[o] = m;
    var[o]  = sq[0] / cnt - m * m;
  }
}

// ---------------- BN stats per-channel over batch for z[rows][C] ----------------
__global__ __launch_bounds__(256) void k_stats_col(const float* __restrict__ z,
                                                   float* __restrict__ mean,
                                                   float* __restrict__ var,
                                                   int rows, int C) {
  __shared__ float ss[256], sq[256];
  int c = blockIdx.x, t = threadIdx.x;
  float s = 0.f, q = 0.f;
  for (int i = t; i < rows; i += 256) {
    float v = z[(size_t)i * C + c];
    s += v; q += v * v;
  }
  ss[t] = s; sq[t] = q; __syncthreads();
  for (int off = 128; off; off >>= 1) {
    if (t < off) { ss[t] += ss[t + off]; sq[t] += sq[t + off]; }
    __syncthreads();
  }
  if (t == 0) {
    float m = ss[0] / (float)rows;
    mean[c] = m;
    var[c]  = sq[0] / (float)rows - m * m;
  }
}

// ---------------- normalize arm -> h bf16 (channel o = (i>>6)&255) ----------------
__global__ void k_norm_arm(const float* __restrict__ arm, const float* __restrict__ mean,
                           const float* __restrict__ var, const float* __restrict__ g,
                           const float* __restrict__ be, bf16_t* __restrict__ h) {
  long i = (long)blockIdx.x * blockDim.x + threadIdx.x;
  if (i >= (long)BSZ * NHID * NEMB) return;
  int o = (int)((i >> 6) & (NHID - 1));
  float v = g[o] * (arm[i] - mean[o]) * rsqrtf(var[o] + EPS) + be[o];
  h[i] = f2bf(v);
}

// ---------------- normalize + relu: z[rows][C] -> bf16 ----------------
__global__ void k_norm_relu(const float* __restrict__ z, const float* __restrict__ mean,
                            const float* __restrict__ var, const float* __restrict__ g,
                            const float* __restrict__ be, bf16_t* __restrict__ h,
                            long total, int C) {
  long i = (long)blockIdx.x * blockDim.x + threadIdx.x;
  if (i >= total) return;
  int c = (int)(i % C);
  float v = g[c] * (z[i] - mean[c]) * rsqrtf(var[c] + EPS) + be[c];
  h[i] = f2bf(fmaxf(v, 0.f));
}

// ---------------- float -> bf16 convert ----------------
__global__ void k_convert(const float* __restrict__ src, bf16_t* __restrict__ dst,
                          long n) {
  long i = (long)blockIdx.x * blockDim.x + threadIdx.x;
  if (i < n) dst[i] = f2bf(src[i]);
}

// ---------------- final: y[b] = h2[b,:] . out_w + out_b ----------------
__global__ __launch_bounds__(256) void k_out(const bf16_t* __restrict__ h2,
                                             const float* __restrict__ wout,
                                             const float* __restrict__ bout,
                                             float* __restrict__ y) {
  __shared__ float ss[256];
  int b = blockIdx.x, t = threadIdx.x;
  float s = 0.f;
  for (int c = t; c < MLP_NHID; c += 256)
    s += bf2f(h2[(size_t)b * MLP_NHID + c]) * wout[c];
  ss[t] = s; __syncthreads();
  for (int off = 128; off; off >>= 1) {
    if (t < off) ss[t] += ss[t + off];
    __syncthreads();
  }
  if (t == 0) y[b] = ss[0] + bout[0];
}

// ======================= host launch =======================
extern "C" void kernel_launch(void* const* d_in, const int* in_sizes, int n_in,
                              void* d_out, int out_size, void* d_ws, size_t ws_size,
                              hipStream_t stream) {
  (void)in_sizes; (void)n_in; (void)out_size; (void)ws_size;
  const int*   ids    = (const int*)d_in[0];
  const float* values = (const float*)d_in[1];
  const float* table  = (const float*)d_in[2];
  const float* w_bil  = (const float*)d_in[3];
  const float* query  = (const float*)d_in[4];
  const float* attv   = (const float*)d_in[5];
  const float* bn_g   = (const float*)d_in[6];
  const float* bn_b   = (const float*)d_in[7];
  const float* w0     = (const float*)d_in[8];
  const float* b0     = (const float*)d_in[9];
  const float* g0     = (const float*)d_in[10];
  const float* be0    = (const float*)d_in[11];
  const float* w1     = (const float*)d_in[12];
  const float* b1     = (const float*)d_in[13];
  const float* g1     = (const float*)d_in[14];
  const float* be1    = (const float*)d_in[15];
  const float* wout   = (const float*)d_in[16];
  const float* bout   = (const float*)d_in[17];
  float* y = (float*)d_out;

  // workspace carve-up (256B aligned)
  uint8_t* ws = (uint8_t*)d_ws;
  size_t off = 0;
  auto carve = [&](size_t bytes) -> uint8_t* {
    uint8_t* p = ws + off;
    off += (bytes + 255) & ~(size_t)255;
    return p;
  };
  bf16_t* emb    = (bf16_t*)carve((size_t)BSZ * NFIELD * NEMB * 2);      // 26 MB
  size_t  embT_b = (size_t)BSZ * NEMB * FPAD * 2;
  bf16_t* embT   = (bf16_t*)carve(embT_b);                               // 34 MB
  bf16_t* wc     = (bf16_t*)carve((size_t)NHID * NEMB * 2);
  bf16_t* armw   = (bf16_t*)carve((size_t)BSZ * NHID * FPAD * 2);        // 134 MB
  float*  arm    = (float*) carve((size_t)BSZ * NHID * NEMB * 4);        // 134 MB
  bf16_t* w0b    = (bf16_t*)carve((size_t)MLP_NHID * NHID * NEMB * 2);   // 34 MB
  bf16_t* w1b    = (bf16_t*)carve((size_t)MLP_NHID * MLP_NHID * 2);      // 2 MB
  bf16_t* h0     = (bf16_t*)carve((size_t)BSZ * NHID * NEMB * 2);        // 67 MB
  float*  z0     = (float*) carve((size_t)BSZ * MLP_NHID * 4);           // 8 MB
  bf16_t* h1     = (bf16_t*)carve((size_t)BSZ * MLP_NHID * 2);
  float*  z1     = (float*) carve((size_t)BSZ * MLP_NHID * 4);
  bf16_t* h2     = (bf16_t*)carve((size_t)BSZ * MLP_NHID * 2);
  float*  meanA  = (float*) carve(NHID * 4);
  float*  varA   = (float*) carve(NHID * 4);
  float*  mean0  = (float*) carve(MLP_NHID * 4);
  float*  var0   = (float*) carve(MLP_NHID * 4);
  float*  mean1  = (float*) carve(MLP_NHID * 4);
  float*  var1   = (float*) carve(MLP_NHID * 4);

  // 0a: combined bilinear+query matrix (scale folded)
  k_wc<<<(NHID * NEMB + 255) / 256, 256, 0, stream>>>(query, w_bil, wc);

  // 0b: zero embT pad, then gather
  (void)hipMemsetAsync(embT, 0, embT_b, stream);
  long ng = (long)BSZ * NFIELD * NEMB;
  k_gather<<<(unsigned)((ng + 255) / 256), 256, 0, stream>>>(ids, values, table, emb, embT);

  // weight conversions
  long n0w = (long)MLP_NHID * NHID * NEMB;
  k_convert<<<(unsigned)((n0w + 255) / 256), 256, 0, stream>>>(w0, w0b, n0w);
  long n1w = (long)MLP_NHID * MLP_NHID;
  k_convert<<<(unsigned)((n1w + 255) / 256), 256, 0, stream>>>(w1, w1b, n1w);

  // gates GEMM + entmax + att weighting (fused)
  k_gates_entmax<<<BSZ * 2, 256, 0, stream>>>(wc, emb, attv, armw);

  // arm = exp(batched GEMM)
  k_arm<<<BSZ, 256, 0, stream>>>(armw, embT, arm);

  // BN over (b,e) per o, normalize -> h0 bf16
  k_stats_arm<<<NHID, 256, 0, stream>>>(arm, meanA, varA);
  long na = (long)BSZ * NHID * NEMB;
  k_norm_arm<<<(unsigned)((na + 255) / 256), 256, 0, stream>>>(arm, meanA, varA, bn_g, bn_b, h0);

  // MLP layer 0: [2048,16384] x [16384,1024]
  k_gemm<<<dim3(BSZ / 128, MLP_NHID / 128), 256, 0, stream>>>(
      h0, w0b, b0, z0, BSZ, MLP_NHID, NHID * NEMB);
  k_stats_col<<<MLP_NHID, 256, 0, stream>>>(z0, mean0, var0, BSZ, MLP_NHID);
  long nz = (long)BSZ * MLP_NHID;
  k_norm_relu<<<(unsigned)((nz + 255) / 256), 256, 0, stream>>>(z0, mean0, var0, g0, be0, h1, nz, MLP_NHID);

  // MLP layer 1: [2048,1024] x [1024,1024]
  k_gemm<<<dim3(BSZ / 128, MLP_NHID / 128), 256, 0, stream>>>(
      h1, w1b, b1, z1, BSZ, MLP_NHID, MLP_NHID);
  k_stats_col<<<MLP_NHID, 256, 0, stream>>>(z1, mean1, var1, BSZ, MLP_NHID);
  k_norm_relu<<<(unsigned)((nz + 255) / 256), 256, 0, stream>>>(z1, mean1, var1, g1, be1, h2, nz, MLP_NHID);

  // output head
  k_out<<<BSZ, 256, 0, stream>>>(h2, wout, bout, y);
}
